// MemoryBank_52759378264646
// MI455X (gfx1250) — matrix operand order repaired and checked by reading, not compiled, not run on hardware
//
#include <hip/hip_runtime.h>
#include <hip/hip_bf16.h>

typedef __attribute__((ext_vector_type(16))) _Float16 v16h;
typedef __attribute__((ext_vector_type(8)))  _Float16 v8h;
typedef __attribute__((ext_vector_type(8)))  float    v8f;

union V16U { v16h v; v8h h[2]; };

#define SPLITS 32
#define TOPK 8
#define NEG_INF (-3.0e38f)

// ---------------- row L2-normalize -> f16 (used for queries AND keys) ----------------
__global__ void rownorm_f16_kernel(const float* __restrict__ x, _Float16* __restrict__ xh, int R) {
    int wid  = (blockIdx.x * blockDim.x + threadIdx.x) >> 5;
    int lane = threadIdx.x & 31;
    if (wid >= R) return;
    const float4 v = ((const float4*)(x + (size_t)wid * 128))[lane];
    float s = v.x*v.x + v.y*v.y + v.z*v.z + v.w*v.w;
    #pragma unroll
    for (int m = 16; m >= 1; m >>= 1) s += __shfl_xor(s, m, 32);
    float inv = 1.0f / fmaxf(sqrtf(s), 1e-12f);
    _Float16* o = xh + (size_t)wid * 128 + lane * 4;
    o[0] = (_Float16)(v.x * inv); o[1] = (_Float16)(v.y * inv);
    o[2] = (_Float16)(v.z * inv); o[3] = (_Float16)(v.w * inv);
}

// ---------------- key inverse norms (fallback path) ----------------
__global__ void knorm_kernel(const float* __restrict__ k, float* __restrict__ invn, int N) {
    int wid  = (blockIdx.x * blockDim.x + threadIdx.x) >> 5;
    int lane = threadIdx.x & 31;
    if (wid >= N) return;
    const float4 v = ((const float4*)(k + (size_t)wid * 128))[lane];
    float s = v.x*v.x + v.y*v.y + v.z*v.z + v.w*v.w;
    #pragma unroll
    for (int m = 16; m >= 1; m >>= 1) s += __shfl_xor(s, m, 32);
    if (lane == 0) invn[wid] = 1.0f / fmaxf(sqrtf(s), 1e-12f);
}

__device__ __forceinline__ void cvt_store8(_Float16* dst, float4 x0, float4 x1, float inv) {
    dst[0] = (_Float16)(x0.x * inv); dst[1] = (_Float16)(x0.y * inv);
    dst[2] = (_Float16)(x0.z * inv); dst[3] = (_Float16)(x0.w * inv);
    dst[4] = (_Float16)(x1.x * inv); dst[5] = (_Float16)(x1.y * inv);
    dst[6] = (_Float16)(x1.z * inv); dst[7] = (_Float16)(x1.w * inv);
}

// Shared top-8 insertion helper (register-resident sorted list, threshold-guarded caller)
__device__ __forceinline__ void top8_insert(float* topS, int* topI, float s, int idx) {
    int j = TOPK - 1;
    while (j > 0 && topS[j - 1] < s) {
        topS[j] = topS[j - 1]; topI[j] = topI[j - 1]; --j;
    }
    topS[j] = s; topI[j] = idx;
}

// ================= PATH A: pre-normalized f16 keys, barrier-free waves =================
// grid: (ceil(B/128), SPLITS), block 256 (8 waves). Wave w owns 16-query tile w and
// streams its split's key tiles with B-fragments loaded straight from global f16 keys.
__global__ __launch_bounds__(256) void gemm_topk_f16_kernel(
    const _Float16* __restrict__ qh, const _Float16* __restrict__ kh,
    float* __restrict__ pscore, int* __restrict__ pidx, int B, int N) {

    const int lane  = threadIdx.x & 31;
    const int w     = threadIdx.x >> 5;
    const int half  = lane >> 4;
    const int l16   = lane & 15;
    const int qb    = blockIdx.x;
    const int split = blockIdx.y;
    const int stride = (int)gridDim.y;

    __shared__ float sims[8][16][20];      // per-wave C transpose, rows 16B-aligned

    const int qrow = qb * 128 + w * 16 + l16;
    const int qr   = qrow < B ? qrow : (B - 1);
    v16h a[4];
    #pragma unroll
    for (int ks = 0; ks < 4; ++ks) {
        const _Float16* base = qh + (size_t)qr * 128 + ks * 32 + half * 8;
        V16U u;
        u.h[0] = *(const v8h*)base;
        u.h[1] = *(const v8h*)(base + 16);
        a[ks] = u.v;
    }

    float topS[TOPK]; int topI[TOPK];
    #pragma unroll
    for (int j = 0; j < TOPK; ++j) { topS[j] = NEG_INF; topI[j] = 0; }

    const int numTiles = (N + 15) >> 4;

    for (int T = split; T < numTiles; T += stride) {
        const int nbase = T * 16;
        int nr = nbase + l16; if (nr > N - 1) nr = N - 1;
        const _Float16* kp = kh + (size_t)nr * 128 + half * 8;

        v16h b[4];
        #pragma unroll
        for (int ks = 0; ks < 4; ++ks) {
            V16U u;
            u.h[0] = *(const v8h*)(kp + ks * 32);
            u.h[1] = *(const v8h*)(kp + ks * 32 + 16);
            b[ks] = u.v;
        }
        int Tp = T + stride;
        if (Tp < numTiles)
            __builtin_prefetch(kh + (size_t)(Tp * 16 + l16) * 128 + half * 64, 0, 1);

        v8f c = {};
        #pragma unroll
        for (int ks = 0; ks < 4; ++ks)
            c = __builtin_amdgcn_wmma_f32_16x16x32_f16(
                    false, a[ks], false, b[ks], (short)0, c, false, false);

        #pragma unroll
        for (int r = 0; r < 8; ++r) sims[w][r + 8 * half][l16] = c[r];
        __builtin_amdgcn_wave_barrier();   // intra-wave: DS pipe is in-order, stop reordering

        if (lane < 16) {
            const float4* rp = (const float4*)&sims[w][lane][0];
            float4 r0 = rp[0], r1 = rp[1], r2 = rp[2], r3 = rp[3];
            float mx = fmaxf(
                fmaxf(fmaxf(fmaxf(r0.x, r0.y), fmaxf(r0.z, r0.w)),
                      fmaxf(fmaxf(r1.x, r1.y), fmaxf(r1.z, r1.w))),
                fmaxf(fmaxf(fmaxf(r2.x, r2.y), fmaxf(r2.z, r2.w)),
                      fmaxf(fmaxf(r3.x, r3.y), fmaxf(r3.z, r3.w))));
            if (mx > topS[TOPK - 1]) {
                float vals[16] = { r0.x, r0.y, r0.z, r0.w, r1.x, r1.y, r1.z, r1.w,
                                   r2.x, r2.y, r2.z, r2.w, r3.x, r3.y, r3.z, r3.w };
                #pragma unroll
                for (int kk = 0; kk < 16; ++kk) {
                    float s = vals[kk];
                    int idx = nbase + kk;
                    if (s > topS[TOPK - 1] && idx < N) top8_insert(topS, topI, s, idx);
                }
            }
        }
        __builtin_amdgcn_wave_barrier();
    }

    if (qrow < B && lane < 16) {
        size_t base = ((size_t)qrow * stride + split) * TOPK;
        #pragma unroll
        for (int j = 0; j < TOPK; ++j) { pscore[base + j] = topS[j]; pidx[base + j] = topI[j]; }
    }
}

// ================= PATH B (fallback, small workspace): staged f32 keys =================
__global__ __launch_bounds__(256) void gemm_topk_staged_kernel(
    const _Float16* __restrict__ qh, const float* __restrict__ keys,
    const float* __restrict__ invn, float* __restrict__ pscore,
    int* __restrict__ pidx, int B, int N) {

    const int lane  = threadIdx.x & 31;
    const int w     = threadIdx.x >> 5;
    const int half  = lane >> 4;
    const int l16   = lane & 15;
    const int qb    = blockIdx.x;
    const int split = blockIdx.y;
    const int stride = (int)gridDim.y;

    __shared__ _Float16 kh[2][16][136];
    __shared__ float sims[8][16][20];

    const int qrow = qb * 128 + w * 16 + l16;
    const int qr   = qrow < B ? qrow : (B - 1);
    v16h a[4];
    #pragma unroll
    for (int ks = 0; ks < 4; ++ks) {
        const _Float16* base = qh + (size_t)qr * 128 + ks * 32 + half * 8;
        V16U u;
        u.h[0] = *(const v8h*)base;
        u.h[1] = *(const v8h*)(base + 16);
        a[ks] = u.v;
    }

    float topS[TOPK]; int topI[TOPK];
    #pragma unroll
    for (int j = 0; j < TOPK; ++j) { topS[j] = NEG_INF; topI[j] = 0; }

    const int numTiles = (N + 15) >> 4;
    const int iters = (numTiles + stride - 1) / stride;
    const int srow = threadIdx.x >> 4;
    const int scol = (threadIdx.x & 15) * 8;

    {
        int T0 = split < numTiles ? split : (numTiles - 1);
        int nc = T0 * 16 + srow; if (nc > N - 1) nc = N - 1;
        float inv = invn[nc];
        const float* kp = keys + (size_t)nc * 128 + scol;
        cvt_store8(&kh[0][srow][scol], ((const float4*)kp)[0], ((const float4*)kp)[1], inv);
    }
    __syncthreads();

    for (int i = 0; i < iters; ++i) {
        const int T = split + i * stride;
        const bool tvalid = (T < numTiles);
        const int nbase = (tvalid ? T : (numTiles - 1)) * 16;
        const int cur = i & 1, nxt = cur ^ 1;

        float4 y0, y1; float ninv;
        {
            int Tn = split + (i + 1) * stride;
            if (Tn > numTiles - 1) Tn = numTiles - 1;
            int nc = Tn * 16 + srow; if (nc > N - 1) nc = N - 1;
            ninv = invn[nc];
            const float* kp = keys + (size_t)nc * 128 + scol;
            y0 = ((const float4*)kp)[0];
            y1 = ((const float4*)kp)[1];
            int Tp = split + (i + 2) * stride;
            if (Tp < numTiles)
                __builtin_prefetch(keys + (size_t)(Tp * 16 + srow) * 128 + scol, 0, 1);
        }

        v16h b[4];
        #pragma unroll
        for (int ks = 0; ks < 4; ++ks) {
            const _Float16* bp = &kh[cur][l16][ks * 32 + half * 8];
            V16U u;
            u.h[0] = *(const v8h*)bp;
            u.h[1] = *(const v8h*)(bp + 16);
            b[ks] = u.v;
        }
        v8f c = {};
        #pragma unroll
        for (int ks = 0; ks < 4; ++ks)
            c = __builtin_amdgcn_wmma_f32_16x16x32_f16(
                    false, a[ks], false, b[ks], (short)0, c, false, false);

        #pragma unroll
        for (int r = 0; r < 8; ++r) sims[w][r + 8 * half][l16] = c[r];
        __syncthreads();

        if (lane < 16) {
            const float4* rp = (const float4*)&sims[w][lane][0];
            float4 r0 = rp[0], r1 = rp[1], r2 = rp[2], r3 = rp[3];
            float mx = fmaxf(
                fmaxf(fmaxf(fmaxf(r0.x, r0.y), fmaxf(r0.z, r0.w)),
                      fmaxf(fmaxf(r1.x, r1.y), fmaxf(r1.z, r1.w))),
                fmaxf(fmaxf(fmaxf(r2.x, r2.y), fmaxf(r2.z, r2.w)),
                      fmaxf(fmaxf(r3.x, r3.y), fmaxf(r3.z, r3.w))));
            if (tvalid && mx > topS[TOPK - 1]) {
                float vals[16] = { r0.x, r0.y, r0.z, r0.w, r1.x, r1.y, r1.z, r1.w,
                                   r2.x, r2.y, r2.z, r2.w, r3.x, r3.y, r3.z, r3.w };
                #pragma unroll
                for (int kk = 0; kk < 16; ++kk) {
                    float s = vals[kk];
                    int idx = nbase + kk;
                    if (s > topS[TOPK - 1] && idx < N) top8_insert(topS, topI, s, idx);
                }
            }
        }

        cvt_store8(&kh[nxt][srow][scol], y0, y1, ninv);
        __syncthreads();
    }

    if (qrow < B && lane < 16) {
        size_t base = ((size_t)qrow * stride + split) * TOPK;
        #pragma unroll
        for (int j = 0; j < TOPK; ++j) { pscore[base + j] = topS[j]; pidx[base + j] = topI[j]; }
    }
}

// ---------------- merge SPLITS partial lists per query ----------------
__global__ void merge_kernel(const float* __restrict__ pscore, const int* __restrict__ pidx,
                             int* __restrict__ fidx, int B, int lists) {
    int b = blockIdx.x * blockDim.x + threadIdx.x;
    if (b >= B) return;
    float topS[TOPK]; int topI[TOPK];
    #pragma unroll
    for (int j = 0; j < TOPK; ++j) { topS[j] = NEG_INF; topI[j] = 0; }
    const float* ps = pscore + (size_t)b * lists * TOPK;
    const int*   pi = pidx   + (size_t)b * lists * TOPK;
    for (int l = 0; l < lists * TOPK; ++l) {
        float s = ps[l];
        if (s > topS[TOPK - 1]) top8_insert(topS, topI, s, pi[l]);
    }
    #pragma unroll
    for (int j = 0; j < TOPK; ++j) fidx[b * TOPK + j] = topI[j];
}

// ---------------- gather value rows, one wave per (b, k) ----------------
__global__ void gather_kernel(const int* __restrict__ fidx, const float* __restrict__ values,
                              float* __restrict__ out, int B) {
    int wid  = (blockIdx.x * blockDim.x + threadIdx.x) >> 5;
    int lane = threadIdx.x & 31;
    if (wid >= B * TOPK) return;
    int idx = fidx[wid];
    float4 v = ((const float4*)(values + (size_t)idx * 128))[lane];
    ((float4*)(out + (size_t)wid * 128))[lane] = v;
}

extern "C" void kernel_launch(void* const* d_in, const int* in_sizes, int n_in,
                              void* d_out, int out_size, void* d_ws, size_t ws_size,
                              hipStream_t stream) {
    const int D = 128;
    const int B = in_sizes[0] / D;
    const int N = in_sizes[1] / D;

    const float* q    = (const float*)d_in[0];
    const float* keys = (const float*)d_in[1];
    const float* vals = (const float*)d_in[2];
    float* out = (float*)d_out;

    char* ws = (char*)d_ws;
    const size_t ALN = 255;
    size_t o = 0;
    _Float16* qh = (_Float16*)(ws + o); o += (size_t)B * D * sizeof(_Float16); o = (o + ALN) & ~ALN;
    float* pscore = (float*)(ws + o);   o += (size_t)B * SPLITS * TOPK * 4;    o = (o + ALN) & ~ALN;
    int*   pidx   = (int*)(ws + o);     o += (size_t)B * SPLITS * TOPK * 4;    o = (o + ALN) & ~ALN;
    int*   fidx   = (int*)(ws + o);     o += (size_t)B * TOPK * 4;             o = (o + ALN) & ~ALN;
    size_t oCommon = o;

    // Path A extra: full f16 normalized key copy
    size_t needA = oCommon + (size_t)N * D * sizeof(_Float16);
    // Path B extra: f32 inverse norms
    size_t needB = oCommon + (size_t)N * sizeof(float);

    dim3 grid((B + 127) / 128, SPLITS);

    rownorm_f16_kernel<<<(B * 32 + 255) / 256, 256, 0, stream>>>(q, qh, B);

    if (ws_size >= needA) {
        _Float16* kh16 = (_Float16*)(ws + oCommon);
        rownorm_f16_kernel<<<(N * 32 + 255) / 256, 256, 0, stream>>>(keys, kh16, N);
        gemm_topk_f16_kernel<<<grid, 256, 0, stream>>>(qh, kh16, pscore, pidx, B, N);
    } else {
        float* invn = (float*)(ws + oCommon);
        knorm_kernel<<<(N * 32 + 255) / 256, 256, 0, stream>>>(keys, invn, N);
        gemm_topk_staged_kernel<<<grid, 256, 0, stream>>>(qh, keys, invn, pscore, pidx, B, N);
    }

    merge_kernel<<<(B + 255) / 256, 256, 0, stream>>>(pscore, pidx, fidx, B, SPLITS);
    gather_kernel<<<(B * TOPK * 32 + 255) / 256, 256, 0, stream>>>(fidx, vals, out, B);
}
